// EncoderLayer_78786880078053
// MI455X (gfx1250) — matrix operand order repaired and checked by reading, not compiled, not run on hardware
//
#include <hip/hip_runtime.h>

#define HEADS  16
#define EMBED  1024
#define DHEAD  64
#define SEQ    1024
#define NBATCH 4
#define FFDIM  4096
#define TTOK   (NBATCH * SEQ)
#define LN_EPS 1e-5f

typedef _Float16 f16;
typedef __attribute__((ext_vector_type(16))) _Float16 v16h;
typedef __attribute__((ext_vector_type(8)))  float    v8f;

// ---------------- WMMA fragment helpers (ISA 7.12.2 layouts, wave32) --------

// A (16xK slice, row-major, 16-bit): lane holds row m=lane&15,
// K offsets {8*half..8*half+7} and {16+8*half..16+8*half+7}.
static __device__ __forceinline__ v16h load_a_frag(const f16* A, int lda, int lane) {
  const int half = lane >> 4;
  const int m    = lane & 15;
  const f16* p = A + (size_t)m * lda + half * 8;
  v16h f;
#pragma unroll
  for (int i = 0; i < 8; ++i) { f[i] = p[i]; f[8 + i] = p[16 + i]; }
  return f;
}

// B (32x16 = K x N) where column n is row n of a row-major (N x K) matrix W
// (i.e. computing X @ W^T). Lane holds col n=lane&15, K = {8*half..8*half+7} and {16+8*half..16+8*half+7}.
static __device__ __forceinline__ v16h load_b_fragT(const f16* W, int ldw, int lane) {
  const int half = lane >> 4;
  const int n    = lane & 15;
  const f16* p = W + (size_t)n * ldw + half * 8;
  v16h f;
#pragma unroll
  for (int i = 0; i < 8; ++i) { f[i] = p[i]; f[8 + i] = p[16 + i]; }
  return f;
}

static __device__ __forceinline__ v8f wmma16(v16h a, v16h b, v8f c) {
  return __builtin_amdgcn_wmma_f32_16x16x32_f16(false, a, false, b, (short)0, c,
                                                false, false);
}

// ---------------- f32 -> f16 convert ----------------------------------------

__global__ void cvt_f32_f16(const float* __restrict__ in, f16* __restrict__ out, int n) {
  int i = blockIdx.x * blockDim.x + threadIdx.x;
  if (i < n) out[i] = (f16)in[i];
}

// ---------------- QKV projection (per-head 64x64 GEMMs) ----------------------
// q,k stored [nb][h][s][d]; v stored transposed [nb][h][d][s].
__global__ void qkv_kernel(const f16* __restrict__ xb,
                           const f16* __restrict__ wq, const f16* __restrict__ wk,
                           const f16* __restrict__ wv,
                           f16* __restrict__ qs, f16* __restrict__ ks,
                           f16* __restrict__ vt) {
  const int lane = threadIdx.x & 31;
  const int wave = threadIdx.x >> 5;
  const int job  = blockIdx.x * 4 + wave;        // 4096 jobs
  const int h    = job & (HEADS - 1);
  const int tile = job >> 4;                     // 0..255  (token tile of 16)
  const int t0   = tile * 16;
  const int nb   = t0 >> 10;
  const int s0   = t0 & (SEQ - 1);
  const int half = lane >> 4, n = lane & 15;

  v16h a0 = load_a_frag(xb + (size_t)t0 * EMBED + h * DHEAD + 0,  EMBED, lane);
  v16h a1 = load_a_frag(xb + (size_t)t0 * EMBED + h * DHEAD + 32, EMBED, lane);

  f16* qh = qs + ((size_t)(nb * HEADS + h) * SEQ) * DHEAD;
  f16* kh = ks + ((size_t)(nb * HEADS + h) * SEQ) * DHEAD;
  f16* vh = vt + ((size_t)(nb * HEADS + h) * DHEAD) * SEQ;
  const f16* Ws[3] = { wq, wk, wv };

#pragma unroll
  for (int w = 0; w < 3; ++w) {
    const f16* W = Ws[w];
#pragma unroll
    for (int nt = 0; nt < 4; ++nt) {
      v8f acc = {};
      acc = wmma16(a0, load_b_fragT(W + nt * 16 * DHEAD + 0,  DHEAD, lane), acc);
      acc = wmma16(a1, load_b_fragT(W + nt * 16 * DHEAD + 32, DHEAD, lane), acc);
#pragma unroll
      for (int r = 0; r < 8; ++r) {
        const int s = s0 + r + 8 * half;
        const int d = nt * 16 + n;
        const f16 v = (f16)acc[r];
        if (w == 0)      qh[(size_t)s * DHEAD + d] = v;
        else if (w == 1) kh[(size_t)s * DHEAD + d] = v;
        else             vh[(size_t)d * SEQ + s]   = v;
      }
    }
  }
}

// ---------------- Flash attention (streaming softmax) ------------------------
__global__ void attn_kernel(const f16* __restrict__ qs, const f16* __restrict__ ks,
                            const f16* __restrict__ vt, const int* __restrict__ mask,
                            f16* __restrict__ ob) {
  __shared__ f16 lds[4][16][32];
  const int lane = threadIdx.x & 31;
  const int wave = threadIdx.x >> 5;
  const int job  = blockIdx.x * 4 + wave;        // 4096 jobs
  const int qt   = job & 63;
  const int h    = (job >> 6) & (HEADS - 1);
  const int nb   = job >> 10;
  const int half = lane >> 4, n = lane & 15;
  const int q0   = qt * 16;

  const f16* qh = qs + ((size_t)(nb * HEADS + h) * SEQ) * DHEAD;
  const f16* kh = ks + ((size_t)(nb * HEADS + h) * SEQ) * DHEAD;
  const f16* vh = vt + ((size_t)(nb * HEADS + h) * DHEAD) * SEQ;
  const int* mk = mask + nb * SEQ;

  const v16h aq0 = load_a_frag(qh + (size_t)q0 * DHEAD + 0,  DHEAD, lane);
  const v16h aq1 = load_a_frag(qh + (size_t)q0 * DHEAD + 32, DHEAD, lane);

  v8f O[4] = {};
  float mrow[8], lrow[8];
#pragma unroll
  for (int r = 0; r < 8; ++r) { mrow[r] = -3.0e38f; lrow[r] = 0.f; }

  const float scale = 0.03125f;                  // 1/sqrt(EMBED)
  for (int j = 0; j < SEQ; j += 32) {
    v8f s0 = {}, s1 = {};
    s0 = wmma16(aq0, load_b_fragT(kh + (size_t)j * DHEAD + 0,         DHEAD, lane), s0);
    s0 = wmma16(aq1, load_b_fragT(kh + (size_t)j * DHEAD + 32,        DHEAD, lane), s0);
    s1 = wmma16(aq0, load_b_fragT(kh + (size_t)(j + 16) * DHEAD + 0,  DHEAD, lane), s1);
    s1 = wmma16(aq1, load_b_fragT(kh + (size_t)(j + 16) * DHEAD + 32, DHEAD, lane), s1);

    const float neg = -3.0e18f;                  // NEG / sqrt(E), effectively -inf
    const bool m0ok = mk[j + n]      != 0;
    const bool m1ok = mk[j + 16 + n] != 0;
#pragma unroll
    for (int r = 0; r < 8; ++r) {
      s0[r] = m0ok ? s0[r] * scale : neg;
      s1[r] = m1ok ? s1[r] * scale : neg;
    }
    // chunk row-max (reduce across the 16 lanes of each half-group)
    float cm[8];
#pragma unroll
    for (int r = 0; r < 8; ++r) cm[r] = fmaxf(s0[r], s1[r]);
#pragma unroll
    for (int m = 1; m <= 8; m <<= 1)
#pragma unroll
      for (int r = 0; r < 8; ++r) cm[r] = fmaxf(cm[r], __shfl_xor(cm[r], m, 32));

    float alpha[8], mnew[8];
#pragma unroll
    for (int r = 0; r < 8; ++r) {
      mnew[r]  = fmaxf(mrow[r], cm[r]);
      alpha[r] = __expf(mrow[r] - mnew[r]);
      mrow[r]  = mnew[r];
    }
    float rs[8];
#pragma unroll
    for (int r = 0; r < 8; ++r) {
      s0[r] = __expf(s0[r] - mnew[r]);
      s1[r] = __expf(s1[r] - mnew[r]);
      rs[r] = s0[r] + s1[r];
    }
#pragma unroll
    for (int m = 1; m <= 8; m <<= 1)
#pragma unroll
      for (int r = 0; r < 8; ++r) rs[r] += __shfl_xor(rs[r], m, 32);
#pragma unroll
    for (int r = 0; r < 8; ++r) lrow[r] = lrow[r] * alpha[r] + rs[r];
#pragma unroll
    for (int t = 0; t < 4; ++t)
#pragma unroll
      for (int r = 0; r < 8; ++r) O[t][r] = O[t][r] * alpha[r];

    // P (16x32) C-layout -> LDS -> A-fragment layout
#pragma unroll
    for (int r = 0; r < 8; ++r) {
      lds[wave][r + 8 * half][n]      = (f16)s0[r];
      lds[wave][r + 8 * half][16 + n] = (f16)s1[r];
    }
    __builtin_amdgcn_wave_barrier();
    asm volatile("s_wait_dscnt 0" ::: "memory");   // CDNA5 split DS counter
    const v16h ap = load_a_frag(&lds[wave][0][0], 32, lane);

#pragma unroll
    for (int t = 0; t < 4; ++t)
      O[t] = wmma16(ap, load_b_fragT(vh + (size_t)(t * 16) * SEQ + j, SEQ, lane), O[t]);
  }

#pragma unroll
  for (int r = 0; r < 8; ++r) {
    const float inv = 1.0f / lrow[r];
    const int tok = nb * SEQ + q0 + r + 8 * half;
#pragma unroll
    for (int t = 0; t < 4; ++t)
      ob[(size_t)tok * EMBED + h * DHEAD + t * 16 + n] = (f16)(O[t][r] * inv);
  }
}

// ---------------- Generic WMMA GEMM:  Y = A @ W^T (+bias)(+resid)(relu) ------
// A: TxK f16 row-major; W: NoutxK f16 row-major.
// 8 waves/block; each wave computes a 32x64 tile (2 M sub-tiles x 4 N sub-tiles)
// so every B fragment is reused by two WMMAs.
__global__ void gemm_kernel(const f16* __restrict__ A, const f16* __restrict__ W,
                            const float* __restrict__ bias, const float* __restrict__ resid,
                            float* __restrict__ outF, f16* __restrict__ outB,
                            int K, int Nout, int relu) {
  const int lane = threadIdx.x & 31;
  const int wid  = threadIdx.x >> 5;             // 0..7 -> N column group
  const int m0    = blockIdx.y * 32;
  const int nbase = blockIdx.x * 512 + wid * 64;
  const int half = lane >> 4, n = lane & 15;

  v8f acc[2][4] = {};
  const f16* Arow0 = A + (size_t)m0 * K;
  const f16* Arow1 = A + (size_t)(m0 + 16) * K;
  for (int kk = 0; kk < K; kk += 32) {
    if (kk + 32 < K) {
      __builtin_prefetch(Arow0 + kk + 32, 0, 1);   // global_prefetch_b8
      __builtin_prefetch(Arow1 + kk + 32, 0, 1);
    }
    const v16h a0 = load_a_frag(Arow0 + kk, K, lane);
    const v16h a1 = load_a_frag(Arow1 + kk, K, lane);
#pragma unroll
    for (int t = 0; t < 4; ++t) {
      const v16h b = load_b_fragT(W + (size_t)(nbase + t * 16) * K + kk, K, lane);
      acc[0][t] = wmma16(a0, b, acc[0][t]);
      acc[1][t] = wmma16(a1, b, acc[1][t]);
    }
  }
#pragma unroll
  for (int rt = 0; rt < 2; ++rt) {
#pragma unroll
    for (int t = 0; t < 4; ++t) {
#pragma unroll
      for (int r = 0; r < 8; ++r) {
        const int row = m0 + rt * 16 + r + 8 * half;
        const int col = nbase + t * 16 + n;
        float v = acc[rt][t][r];
        if (bias)  v += bias[col];
        if (resid) v += resid[(size_t)row * Nout + col];
        if (relu)  v = fmaxf(v, 0.f);
        if (outF) outF[(size_t)row * Nout + col] = v;
        if (outB) outB[(size_t)row * Nout + col] = (f16)v;
      }
    }
  }
}

// ---------------- LayerNorm (one token per block, E=1024) --------------------
__global__ void ln_kernel(const float* __restrict__ x, const float* __restrict__ g,
                          const float* __restrict__ b, float* __restrict__ outF,
                          f16* __restrict__ outB) {
  __shared__ float red[256];
  const int t = blockIdx.x;
  const int tid = threadIdx.x;
  const float* xr = x + (size_t)t * EMBED;
  float v[4];
  float s = 0.f, ss = 0.f;
#pragma unroll
  for (int i = 0; i < 4; ++i) { v[i] = xr[tid + i * 256]; s += v[i]; ss += v[i] * v[i]; }
  red[tid] = s; __syncthreads();
  for (int o = 128; o > 0; o >>= 1) { if (tid < o) red[tid] += red[tid + o]; __syncthreads(); }
  const float mean = red[0] * (1.0f / EMBED);
  __syncthreads();
  red[tid] = ss; __syncthreads();
  for (int o = 128; o > 0; o >>= 1) { if (tid < o) red[tid] += red[tid + o]; __syncthreads(); }
  const float var  = red[0] * (1.0f / EMBED) - mean * mean;
  const float rstd = rsqrtf(var + LN_EPS);
#pragma unroll
  for (int i = 0; i < 4; ++i) {
    const int e = tid + i * 256;
    const float y = (v[i] - mean) * rstd * g[e] + b[e];
    if (outF) outF[(size_t)t * EMBED + e] = y;
    if (outB) outB[(size_t)t * EMBED + e] = (f16)y;
  }
}

// ---------------- Channel exchange -------------------------------------------
__global__ void exchange_kernel(const float* __restrict__ y0, const float* __restrict__ y1,
                                const float* __restrict__ g2a, const float* __restrict__ g2b,
                                float* __restrict__ z0, float* __restrict__ z1) {
  const size_t i = (size_t)blockIdx.x * blockDim.x + threadIdx.x;
  if (i >= (size_t)TTOK * EMBED) return;
  const int e = (int)(i & (EMBED - 1));
  const float a = y0[i], b = y1[i];
  z0[i] = (fabsf(g2a[e]) >= 0.5f) ? a : b;
  z1[i] = (fabsf(g2b[e]) >= 0.5f) ? b : a;
}

// ---------------- host driver -------------------------------------------------
extern "C" void kernel_launch(void* const* d_in, const int* in_sizes, int n_in,
                              void* d_out, int out_size, void* d_ws, size_t ws_size,
                              hipStream_t stream) {
  (void)in_sizes; (void)n_in; (void)out_size; (void)ws_size;
  const float* x0  = (const float*)d_in[0];
  const float* x1  = (const float*)d_in[1];
  const int*   msk = (const int*)  d_in[2];
  const float* Wq  = (const float*)d_in[3];
  const float* Wk  = (const float*)d_in[4];
  const float* Wv  = (const float*)d_in[5];
  const float* Wo  = (const float*)d_in[6];
  const float* bo  = (const float*)d_in[7];
  const float* W1  = (const float*)d_in[8];
  const float* b1  = (const float*)d_in[9];
  const float* W2  = (const float*)d_in[10];
  const float* b2  = (const float*)d_in[11];
  const float* g1a = (const float*)d_in[12];
  const float* be1a= (const float*)d_in[13];
  const float* g1b = (const float*)d_in[14];
  const float* be1b= (const float*)d_in[15];
  const float* g2a = (const float*)d_in[16];
  const float* g2b = (const float*)d_in[17];

  char* ws = (char*)d_ws;
  size_t off = 0;
  auto alloc = [&](size_t bytes) -> char* {
    char* p = ws + off;
    off += (bytes + 255) & ~size_t(255);
    return p;
  };
  const size_t TE = (size_t)TTOK * EMBED;
  f16*   XB  = (f16*)  alloc(TE * 2);
  f16*   QS  = (f16*)  alloc(TE * 2);
  f16*   KS  = (f16*)  alloc(TE * 2);
  f16*   VT  = (f16*)  alloc(TE * 2);
  f16*   OB  = (f16*)  alloc(TE * 2);
  f16*   FFH = QS;                       // overlay: q/k/v/o dead before FF1
  float* PRE = (float*)alloc(TE * 4);
  float* HF  = (float*)alloc(TE * 4);
  f16*   HB  = (f16*)  alloc(TE * 2);
  float* Y0  = (float*)alloc(TE * 4);
  float* Y1  = (float*)alloc(TE * 4);
  f16*   WQB = (f16*)  alloc(DHEAD * DHEAD * 2);
  f16*   WKB = (f16*)  alloc(DHEAD * DHEAD * 2);
  f16*   WVB = (f16*)  alloc(DHEAD * DHEAD * 2);
  f16*   WOB = (f16*)  alloc((size_t)EMBED * EMBED * 2);
  f16*   W1B = (f16*)  alloc((size_t)FFDIM * EMBED * 2);
  f16*   W2B = (f16*)  alloc((size_t)EMBED * FFDIM * 2);

  auto cvt = [&](const float* src, f16* dst, int n) {
    cvt_f32_f16<<<(n + 255) / 256, 256, 0, stream>>>(src, dst, n);
  };
  cvt(Wq, WQB, DHEAD * DHEAD);
  cvt(Wk, WKB, DHEAD * DHEAD);
  cvt(Wv, WVB, DHEAD * DHEAD);
  cvt(Wo, WOB, EMBED * EMBED);
  cvt(W1, W1B, FFDIM * EMBED);
  cvt(W2, W2B, EMBED * FFDIM);

  auto run_branch = [&](const float* x, const float* g1, const float* be1, float* Y) {
    cvt(x, XB, (int)TE);
    qkv_kernel<<<1024, 128, 0, stream>>>(XB, WQB, WKB, WVB, QS, KS, VT);
    attn_kernel<<<1024, 128, 0, stream>>>(QS, KS, VT, msk, OB);
    // attn_out @ Wo^T + bo + x
    gemm_kernel<<<dim3(EMBED / 512, TTOK / 32), 256, 0, stream>>>(
        OB, WOB, bo, x, PRE, (f16*)nullptr, EMBED, EMBED, 0);
    ln_kernel<<<TTOK, 256, 0, stream>>>(PRE, g1, be1, HF, HB);
    // FF1: relu(h @ W1^T + b1)
    gemm_kernel<<<dim3(FFDIM / 512, TTOK / 32), 256, 0, stream>>>(
        HB, W1B, b1, (const float*)nullptr, (float*)nullptr, FFH, EMBED, FFDIM, 1);
    // FF2: ffh @ W2^T + b2 + h
    gemm_kernel<<<dim3(EMBED / 512, TTOK / 32), 256, 0, stream>>>(
        FFH, W2B, b2, HF, PRE, (f16*)nullptr, FFDIM, EMBED, 0);
    ln_kernel<<<TTOK, 256, 0, stream>>>(PRE, g1, be1, Y, (f16*)nullptr);
  };
  run_branch(x0, g1a, be1a, Y0);
  run_branch(x1, g1b, be1b, Y1);

  float* z0 = (float*)d_out;
  float* z1 = z0 + TE;
  exchange_kernel<<<(int)((TE + 255) / 256), 256, 0, stream>>>(Y0, Y1, g2a, g2b, z0, z1);
}
